// Network_47502338294626
// MI455X (gfx1250) — matrix operand order repaired and checked by reading, not compiled, not run on hardware
//
#include <hip/hip_runtime.h>
#include <hip/hip_bf16.h>

typedef __attribute__((ext_vector_type(16))) _Float16 v16h;
typedef __attribute__((ext_vector_type(8)))  _Float16 v8h;
typedef __attribute__((ext_vector_type(8)))  float    v8f;

union V16 { v16h v; v8h h[2]; };

// ---------------------------------------------------------------------------
// Prep kernel: f32->f16 padded weight copies + pre-reduced W*r sums.
// Padded layouts (row-major, [Npad, Kpad]):
//   decW1h/decW2h : [144,160], dec4W1h/dec4W2h : [64,64]
// ---------------------------------------------------------------------------
#define SEG0 (144*160)          // decW1h
#define SEG1 (SEG0 + 144*160)   // decW2h
#define SEG2 (SEG1 + 64*64)     // dec4W1h
#define SEG3 (SEG2 + 64*64)     // dec4W2h
#define SEG4 (SEG3 + 130)       // W1r
#define SEG5 (SEG4 + 70)        // W2r
#define SEG6 (SEG5 + 50)        // W4r
#define SEG7 (SEG6 + 2)         // W6r
#define PREP_TOT SEG7

__global__ void prep_kernel(const float* __restrict__ decW1, const float* __restrict__ decW2,
                            const float* __restrict__ dec4W1, const float* __restrict__ dec4W2,
                            const float* __restrict__ W1, const float* __restrict__ W2,
                            const float* __restrict__ W4, const float* __restrict__ W6,
                            _Float16* decW1h, _Float16* decW2h,
                            _Float16* d4W1h, _Float16* d4W2h,
                            float* W1r, float* W2r, float* W4r, float* W6r) {
  int i = blockIdx.x * blockDim.x + threadIdx.x;
  if (i < SEG0) {
    int n = i / 160, k = i % 160;
    decW1h[i] = (_Float16)((n < 130 && k < 130) ? decW1[n * 130 + k] : 0.f);
  } else if (i < SEG1) {
    int j = i - SEG0; int n = j / 160, k = j % 160;
    decW2h[j] = (_Float16)((n < 130 && k < 130) ? decW2[n * 130 + k] : 0.f);
  } else if (i < SEG2) {
    int j = i - SEG1; int n = j / 64, k = j % 64;
    d4W1h[j] = (_Float16)((n < 50 && k < 50) ? dec4W1[n * 50 + k] : 0.f);
  } else if (i < SEG3) {
    int j = i - SEG2; int n = j / 64, k = j % 64;
    d4W2h[j] = (_Float16)((n < 50 && k < 50) ? dec4W2[n * 50 + k] : 0.f);
  } else if (i < SEG4) {
    int j = i - SEG3; float s = 0.f;
    for (int q = 0; q < 10; ++q) s += W1[j * 10 + q];
    W1r[j] = s;
  } else if (i < SEG5) {
    int j = i - SEG4; float s = 0.f;
    for (int q = 0; q < 10; ++q) s += W2[j * 10 + q];
    W2r[j] = s;
  } else if (i < SEG6) {
    int j = i - SEG5; float s = 0.f;
    for (int q = 0; q < 10; ++q) s += W4[j * 10 + q];
    W4r[j] = s;
  } else if (i < SEG7) {
    int j = i - SEG6; float s = 0.f;
    for (int q = 0; q < 10; ++q) s += W6[j * 10 + q];
    W6r[j] = s;
  }
}

// ---------------------------------------------------------------------------
// Chain kernel: one thread per batch element; small weights staged in LDS.
// ---------------------------------------------------------------------------
struct ChainArgs {
  const float *x, *w0, *b0, *E1, *E2, *E4, *E6;
  const float *b1, *W2, *b2, *W3, *b3, *W4, *b4, *W5, *b5, *W6, *b6;
  const float *od2W, *od2b, *od40W, *od40b, *od41W, *od41b, *od42W, *od42b;
  const float *W1r, *W2r, *W4r, *W6r;
  _Float16 *A1h, *A4h;
  float *out;
  int B;
};

// LDS float offsets
#define O_E1   0
#define O_E2   169
#define O_E4   260
#define O_E6   325
#define O_W0   351
#define O_B0   364
#define O_W1R  377
#define O_B1   507
#define O_W2   637
#define O_W2R  1337
#define O_B2   1407
#define O_W3   1477
#define O_B3   2477
#define O_W4   2577
#define O_W4R  3077
#define O_B4   3127
#define O_W5   3177
#define O_B5   4177
#define O_W6   4277
#define O_W6R  4297
#define O_B6   4299
#define O_OD2W 4301
#define O_OD2B 4311
#define O_OD40W 4312
#define O_OD40B 4322
#define O_OD41W 4323
#define O_OD41B 4333
#define O_OD42W 4334
#define O_OD42B 4344
#define LDS_TOT 4345

__device__ __forceinline__ void cpylds(float* dst, const float* src, int n, int tid, int nt) {
  for (int i = tid; i < n; i += nt) dst[i] = src[i];
}
__device__ __forceinline__ float lrelu(float v) { return v > 0.f ? v : 0.01f * v; }

__global__ __launch_bounds__(256)
void chain_kernel(ChainArgs a) {
  __shared__ float sm[LDS_TOT];
  const int tid = threadIdx.x, nt = blockDim.x;
  cpylds(sm + O_E1, a.E1, 169, tid, nt);
  cpylds(sm + O_E2, a.E2, 91, tid, nt);
  cpylds(sm + O_E4, a.E4, 65, tid, nt);
  cpylds(sm + O_E6, a.E6, 26, tid, nt);
  cpylds(sm + O_W0, a.w0, 13, tid, nt);
  cpylds(sm + O_B0, a.b0, 13, tid, nt);
  cpylds(sm + O_W1R, a.W1r, 130, tid, nt);
  cpylds(sm + O_B1, a.b1, 130, tid, nt);
  cpylds(sm + O_W2, a.W2, 700, tid, nt);
  cpylds(sm + O_W2R, a.W2r, 70, tid, nt);
  cpylds(sm + O_B2, a.b2, 70, tid, nt);
  cpylds(sm + O_W3, a.W3, 1000, tid, nt);
  cpylds(sm + O_B3, a.b3, 100, tid, nt);
  cpylds(sm + O_W4, a.W4, 500, tid, nt);
  cpylds(sm + O_W4R, a.W4r, 50, tid, nt);
  cpylds(sm + O_B4, a.b4, 50, tid, nt);
  cpylds(sm + O_W5, a.W5, 1000, tid, nt);
  cpylds(sm + O_B5, a.b5, 100, tid, nt);
  cpylds(sm + O_W6, a.W6, 20, tid, nt);
  cpylds(sm + O_W6R, a.W6r, 2, tid, nt);
  cpylds(sm + O_B6, a.b6, 2, tid, nt);
  cpylds(sm + O_OD2W, a.od2W, 10, tid, nt);
  cpylds(sm + O_OD2B, a.od2b, 1, tid, nt);
  cpylds(sm + O_OD40W, a.od40W, 10, tid, nt);
  cpylds(sm + O_OD40B, a.od40b, 1, tid, nt);
  cpylds(sm + O_OD41W, a.od41W, 10, tid, nt);
  cpylds(sm + O_OD41B, a.od41b, 1, tid, nt);
  cpylds(sm + O_OD42W, a.od42W, 10, tid, nt);
  cpylds(sm + O_OD42B, a.od42b, 1, tid, nt);
  __syncthreads();

  const int b = blockIdx.x * blockDim.x + threadIdx.x;
  if (b >= a.B) return;
  const size_t Bt = (size_t)a.B;
  float* outp = a.out;
  float* l1in = outp + 136 * Bt + (size_t)b * 130;   // l1_in_flat
  float* l4in = outp + 316 * Bt + (size_t)b * 50;    // l4_in_flat
  float* outv = outp + (size_t)b * 6;                // output

  // layer0
  float o0[13];
#pragma unroll
  for (int s = 0; s < 13; ++s)
    o0[s] = a.x[(size_t)b * 13 + s] * sm[O_W0 + s] + sm[O_B0 + s];

  float s1v[13], s2v[7], s4v[5], s6v[2];
#pragma unroll
  for (int n = 0; n < 13; ++n) { float t = 0.f;
#pragma unroll
    for (int s = 0; s < 13; ++s) t += o0[s] * sm[O_E1 + s * 13 + n];
    s1v[n] = t; }
#pragma unroll
  for (int n = 0; n < 7; ++n) { float t = 0.f;
#pragma unroll
    for (int s = 0; s < 13; ++s) t += o0[s] * sm[O_E2 + s * 7 + n];
    s2v[n] = t; }
#pragma unroll
  for (int n = 0; n < 5; ++n) { float t = 0.f;
#pragma unroll
    for (int s = 0; s < 13; ++s) t += o0[s] * sm[O_E4 + s * 5 + n];
    s4v[n] = t; }
#pragma unroll
  for (int n = 0; n < 2; ++n) { float t = 0.f;
#pragma unroll
    for (int s = 0; s < 13; ++s) t += o0[s] * sm[O_E6 + s * 2 + n];
    s6v[n] = t; }

  // layer1 (broadcast scalar input -> pre-reduced W1r)
  _Float16* A1 = a.A1h + (size_t)b * 160;
  float sum1[10];
#pragma unroll
  for (int o = 0; o < 10; ++o) sum1[o] = 0.f;
#pragma unroll
  for (int n = 0; n < 13; ++n) {
    float sn = s1v[n];
#pragma unroll
    for (int o = 0; o < 10; ++o) {
      float t = lrelu(sn * sm[O_W1R + n * 10 + o] + sm[O_B1 + n * 10 + o]);
      sum1[o] += t;
      A1[n * 10 + o] = (_Float16)t;
      l1in[n * 10 + o] = sn;
    }
  }
#pragma unroll
  for (int k = 130; k < 160; ++k) A1[k] = (_Float16)0.f;

  // layer2
  float sum2[10], o2r3[10];
#pragma unroll
  for (int o = 0; o < 10; ++o) sum2[o] = 0.f;
#pragma unroll
  for (int n = 0; n < 7; ++n) {
    float sn = s2v[n];
#pragma unroll
    for (int o = 0; o < 10; ++o) {
      float t = sn * sm[O_W2R + n * 10 + o] + sm[O_B2 + n * 10 + o];
#pragma unroll
      for (int i = 0; i < 10; ++i) t += sum1[i] * sm[O_W2 + (n * 10 + o) * 10 + i];
      t = lrelu(t);
      sum2[o] += t;
      if (n == 3) o2r3[o] = t;
    }
  }

  // layer3
  float sum3[10];
#pragma unroll
  for (int o = 0; o < 10; ++o) sum3[o] = 0.f;
#pragma unroll
  for (int n = 0; n < 10; ++n) {
#pragma unroll
    for (int o = 0; o < 10; ++o) {
      float t = sm[O_B3 + n * 10 + o];
#pragma unroll
      for (int i = 0; i < 10; ++i) t += sum2[i] * sm[O_W3 + (n * 10 + o) * 10 + i];
      sum3[o] += lrelu(t);
    }
  }

  // layer4
  _Float16* A4 = a.A4h + (size_t)b * 64;
  float sum4[10], o4r[3][10];
#pragma unroll
  for (int o = 0; o < 10; ++o) sum4[o] = 0.f;
#pragma unroll
  for (int n = 0; n < 5; ++n) {
    float sn = s4v[n];
#pragma unroll
    for (int i = 0; i < 10; ++i) l4in[n * 10 + i] = sn + sum3[i];
#pragma unroll
    for (int o = 0; o < 10; ++o) {
      float t = sn * sm[O_W4R + n * 10 + o] + sm[O_B4 + n * 10 + o];
#pragma unroll
      for (int i = 0; i < 10; ++i) t += sum3[i] * sm[O_W4 + (n * 10 + o) * 10 + i];
      t = lrelu(t);
      sum4[o] += t;
      A4[n * 10 + o] = (_Float16)t;
      if (n < 3) o4r[n][o] = t;
    }
  }
#pragma unroll
  for (int k = 50; k < 64; ++k) A4[k] = (_Float16)0.f;

  // layer5
  float sum5[10];
#pragma unroll
  for (int o = 0; o < 10; ++o) sum5[o] = 0.f;
#pragma unroll
  for (int n = 0; n < 10; ++n) {
#pragma unroll
    for (int o = 0; o < 10; ++o) {
      float t = sm[O_B5 + n * 10 + o];
#pragma unroll
      for (int i = 0; i < 10; ++i) t += sum4[i] * sm[O_W5 + (n * 10 + o) * 10 + i];
      sum5[o] += lrelu(t);
    }
  }

  // layer6 -> output cols 4,5
#pragma unroll
  for (int n = 0; n < 2; ++n) {
    float t = s6v[n] * sm[O_W6R + n] + sm[O_B6 + n];
#pragma unroll
    for (int i = 0; i < 10; ++i) t += sum5[i] * sm[O_W6 + n * 10 + i];
    outv[4 + n] = lrelu(t);
  }

  // output decoder heads -> cols 0..3
  {
    float t = sm[O_OD2B];
#pragma unroll
    for (int i = 0; i < 10; ++i) t += o2r3[i] * sm[O_OD2W + i];
    outv[0] = t > 0.f ? t : 0.f;
  }
  {
    float t = sm[O_OD40B];
#pragma unroll
    for (int i = 0; i < 10; ++i) t += o4r[0][i] * sm[O_OD40W + i];
    outv[1] = t > 0.f ? t : 0.f;
  }
  {
    float t = sm[O_OD41B];
#pragma unroll
    for (int i = 0; i < 10; ++i) t += o4r[1][i] * sm[O_OD41W + i];
    outv[2] = t > 0.f ? t : 0.f;
  }
  {
    float t = sm[O_OD42B];
#pragma unroll
    for (int i = 0; i < 10; ++i) t += o4r[2][i] * sm[O_OD42W + i];
    outv[3] = t > 0.f ? t : 0.f;
  }
}

// ---------------------------------------------------------------------------
// Fused 2-stage decoder GEMM: out = relu(A @ W1^T + b1) @ W2^T + b2
// Block = NWAVES waves (one 16-wide N tile each) x MT m-tiles of 16 rows.
// W1/W2 tiles staged in LDS once per block; A tile staged per m-tile; hidden
// tile lives in LDS. All WMMA operands come from LDS (ds_load_b128).
// Dynamic LDS layout (halves): Ws1[NW*16*KP] | Ws2[NW*16*KP] | As[16*KP] | Hs[16*KP]
// ---------------------------------------------------------------------------
extern __shared__ __align__(16) _Float16 ldsbuf[];

template <int NWAVES, int KCH, int KP, int NVALID, int MT>
__global__ __launch_bounds__(NWAVES * 32)
void decoder_fused(const _Float16* __restrict__ A,
                   const _Float16* __restrict__ W1h, const float* __restrict__ bias1,
                   const _Float16* __restrict__ W2h, const float* __restrict__ bias2,
                   float* __restrict__ outp) {
  constexpr int WELT = NWAVES * 16 * KP;   // halves per weight matrix
  _Float16* Ws1 = ldsbuf;
  _Float16* Ws2 = Ws1 + WELT;
  _Float16* As  = Ws2 + WELT;
  _Float16* Hs  = As + 16 * KP;

  const int tid  = threadIdx.x;
  const int nt   = NWAVES * 32;
  const int wave = tid >> 5;
  const int lane = tid & 31;
  const bool hi  = lane >= 16;
  const int col  = lane & 15;
  const int n0   = wave * 16;
  const int nglob = n0 + col;

  // stage weights into LDS once per block (contiguous 128-bit copies)
  {
    const v8h* s1 = (const v8h*)W1h;
    const v8h* s2 = (const v8h*)W2h;
    v8h* d1 = (v8h*)Ws1;
    v8h* d2 = (v8h*)Ws2;
    for (int i = tid; i < WELT / 8; i += nt) { d1[i] = s1[i]; d2[i] = s2[i]; }
  }
  // zero the K-pad columns of the hidden tile (stage A never writes them)
  constexpr int PADW = KP - NWAVES * 16;
  if constexpr (PADW > 0) {
    for (int i = tid; i < 16 * PADW; i += nt) {
      int r = i / PADW, c = i % PADW;
      Hs[r * KP + NWAVES * 16 + c] = (_Float16)0.f;
    }
  }
  const float bA = (nglob < NVALID) ? bias1[nglob] : 0.f;
  const float bB = (nglob < NVALID) ? bias2[nglob] : 0.f;
  __syncthreads();

  const int rowW = (n0 + col) * KP;

  for (int mt = 0; mt < MT; ++mt) {
    const int m0 = (blockIdx.x * MT + mt) * 16;

    // cooperative copy of the contiguous 16-row A tile into LDS
    {
      const v8h* src = (const v8h*)(A + (size_t)m0 * KP);
      v8h* dst = (v8h*)As;
      for (int i = tid; i < 16 * KP / 8; i += nt) dst[i] = src[i];
    }
    __syncthreads();

    // ---- stage A: H = relu(A @ W1^T + b1) ----
    v8f acc = {};
#pragma unroll
    for (int kc = 0; kc < KCH; ++kc) {
      const int k0 = kc * 32;
      V16 av, bv;
      av.h[0] = *(const v8h*)(As + col * KP + k0 + (hi ? 8 : 0));
      av.h[1] = *(const v8h*)(As + col * KP + k0 + 16 + (hi ? 8 : 0));
      bv.h[0] = *(const v8h*)(Ws1 + rowW + k0 + (hi ? 8 : 0));
      bv.h[1] = *(const v8h*)(Ws1 + rowW + k0 + 16 + (hi ? 8 : 0));
      acc = __builtin_amdgcn_wmma_f32_16x16x32_f16(false, av.v, false, bv.v,
                                                   (short)0, acc, false, false);
    }
#pragma unroll
    for (int v = 0; v < 8; ++v) {
      const int m = v + (hi ? 8 : 0);
      float val = acc[v] + bA;
      val = val > 0.f ? val : 0.f;
      if (nglob >= NVALID) val = 0.f;   // keep pad columns exactly zero
      Hs[m * KP + nglob] = (_Float16)val;
    }
    __syncthreads();

    // ---- stage B: out = H @ W2^T + b2 ----
    v8f acc2 = {};
#pragma unroll
    for (int kc = 0; kc < KCH; ++kc) {
      const int k0 = kc * 32;
      V16 av, bv;
      av.h[0] = *(const v8h*)(Hs + col * KP + k0 + (hi ? 8 : 0));
      av.h[1] = *(const v8h*)(Hs + col * KP + k0 + 16 + (hi ? 8 : 0));
      bv.h[0] = *(const v8h*)(Ws2 + rowW + k0 + (hi ? 8 : 0));
      bv.h[1] = *(const v8h*)(Ws2 + rowW + k0 + 16 + (hi ? 8 : 0));
      acc2 = __builtin_amdgcn_wmma_f32_16x16x32_f16(false, av.v, false, bv.v,
                                                    (short)0, acc2, false, false);
    }
    if (nglob < NVALID) {
#pragma unroll
      for (int v = 0; v < 8; ++v) {
        const int m = m0 + v + (hi ? 8 : 0);
        outp[(size_t)m * NVALID + nglob] = acc2[v] + bB;
      }
    }
    __syncthreads();  // protect As/Hs before next iteration overwrites them
  }
}

// ---------------------------------------------------------------------------
// Launch
// ---------------------------------------------------------------------------
extern "C" void kernel_launch(void* const* d_in, const int* in_sizes, int n_in,
                              void* d_out, int out_size, void* d_ws, size_t ws_size,
                              hipStream_t stream) {
  const float* x     = (const float*)d_in[0];
  const float* w0    = (const float*)d_in[1];
  const float* b0    = (const float*)d_in[2];
  const float* E1    = (const float*)d_in[3];
  const float* E2    = (const float*)d_in[4];
  const float* E4    = (const float*)d_in[5];
  const float* E6    = (const float*)d_in[6];
  const float* W1    = (const float*)d_in[7];
  const float* b1    = (const float*)d_in[8];
  const float* W2    = (const float*)d_in[9];
  const float* b2    = (const float*)d_in[10];
  const float* W3    = (const float*)d_in[11];
  const float* b3    = (const float*)d_in[12];
  const float* W4    = (const float*)d_in[13];
  const float* b4    = (const float*)d_in[14];
  const float* W5    = (const float*)d_in[15];
  const float* b5    = (const float*)d_in[16];
  const float* W6    = (const float*)d_in[17];
  const float* b6    = (const float*)d_in[18];
  const float* decW1 = (const float*)d_in[19];
  const float* decb1 = (const float*)d_in[20];
  const float* decW2 = (const float*)d_in[21];
  const float* decb2 = (const float*)d_in[22];
  const float* dec4W1= (const float*)d_in[23];
  const float* dec4b1= (const float*)d_in[24];
  const float* dec4W2= (const float*)d_in[25];
  const float* dec4b2= (const float*)d_in[26];
  const float* od2W  = (const float*)d_in[27];
  const float* od2b  = (const float*)d_in[28];
  const float* od40W = (const float*)d_in[29];
  const float* od40b = (const float*)d_in[30];
  const float* od41W = (const float*)d_in[31];
  const float* od41b = (const float*)d_in[32];
  const float* od42W = (const float*)d_in[33];
  const float* od42b = (const float*)d_in[34];

  const int B = in_sizes[0] / 13;
  float* outf = (float*)d_out;

  // workspace layout
  char* ws = (char*)d_ws;
  size_t off = 0;
  _Float16* A1h    = (_Float16*)(ws + off); off += (size_t)B * 160 * 2;
  _Float16* A4h    = (_Float16*)(ws + off); off += (size_t)B * 64 * 2;
  _Float16* decW1h = (_Float16*)(ws + off); off += 144 * 160 * 2;
  _Float16* decW2h = (_Float16*)(ws + off); off += 144 * 160 * 2;
  _Float16* d4W1h  = (_Float16*)(ws + off); off += 64 * 64 * 2;
  _Float16* d4W2h  = (_Float16*)(ws + off); off += 64 * 64 * 2;
  float* W1r = (float*)(ws + off); off += 130 * 4;
  float* W2r = (float*)(ws + off); off += 70 * 4;
  float* W4r = (float*)(ws + off); off += 50 * 4;
  float* W6r = (float*)(ws + off); off += 2 * 4;

  prep_kernel<<<(PREP_TOT + 255) / 256, 256, 0, stream>>>(
      decW1, decW2, dec4W1, dec4W2, W1, W2, W4, W6,
      decW1h, decW2h, d4W1h, d4W2h, W1r, W2r, W4r, W6r);

  ChainArgs ca;
  ca.x = x; ca.w0 = w0; ca.b0 = b0; ca.E1 = E1; ca.E2 = E2; ca.E4 = E4; ca.E6 = E6;
  ca.b1 = b1; ca.W2 = W2; ca.b2 = b2; ca.W3 = W3; ca.b3 = b3; ca.W4 = W4; ca.b4 = b4;
  ca.W5 = W5; ca.b5 = b5; ca.W6 = W6; ca.b6 = b6;
  ca.od2W = od2W; ca.od2b = od2b; ca.od40W = od40W; ca.od40b = od40b;
  ca.od41W = od41W; ca.od41b = od41b; ca.od42W = od42W; ca.od42b = od42b;
  ca.W1r = W1r; ca.W2r = W2r; ca.W4r = W4r; ca.W6r = W6r;
  ca.A1h = A1h; ca.A4h = A4h; ca.out = outf; ca.B = B;
  chain_kernel<<<(B + 255) / 256, 256, 0, stream>>>(ca);

  float* decoded  = outf + (size_t)6 * B;    // [B,130]
  float* decoded4 = outf + (size_t)266 * B;  // [B,50]

  // dynamic LDS sizes (bytes): 2 weight tiles + A tile + H tile
  const size_t sh130 = (size_t)(2 * 144 * 160 + 2 * 16 * 160) * 2;  // 102400 B
  const size_t sh50  = (size_t)(2 * 64 * 64 + 2 * 16 * 64) * 2;     //  20480 B

  decoder_fused<9, 5, 160, 130, 8><<<B / 128, 9 * 32, sh130, stream>>>(
      A1h, decW1h, decb1, decW2h, decb2, decoded);
  decoder_fused<4, 2, 64, 50, 8><<<B / 128, 4 * 32, sh50, stream>>>(
      A4h, d4W1h, dec4b1, d4W2h, dec4b2, decoded4);
}
